// LFSR_72370198938123
// MI455X (gfx1250) — matrix operand order repaired and checked by reading, not compiled, not run on hardware
//
#include <hip/hip_runtime.h>
#include <hip/hip_bf16.h>

// ---------------------------------------------------------------------------
// LFSR network on gfx1250: bf16 WMMA implicit-GEMM convs, f32 accumulation.
// Activation layout: [img][98][98][C] bf16, zero halo -> no boundary checks.
// K-loops fully unrolled: all view/tap/channel decomposition constant-folds,
// A/B fragment loads become immediate-offset global_load_b128.
// ---------------------------------------------------------------------------

typedef __attribute__((ext_vector_type(16))) __bf16       v16bf;
typedef __attribute__((ext_vector_type(8)))  float        f32x8;
typedef __attribute__((ext_vector_type(4)))  unsigned int u32x4;

union FragAB { u32x4 q[2]; v16bf v; };

__device__ __forceinline__ unsigned short f2bf(float f) {
    unsigned int u = __float_as_uint(f);
    u += 0x7fffu + ((u >> 16) & 1u);          // round-to-nearest-even
    return (unsigned short)(u >> 16);
}

// ------------------------------- conv modes --------------------------------
enum { M_SPATIAL = 0, M_ANGULAR = 1, M_VSNL = 2, M_VRN2 = 3, M_VRN3 = 4 };

template <int MODE>
__device__ __forceinline__ int in_img_idx(int o, int v) {
    if constexpr (MODE == M_SPATIAL) { (void)v; return o; }
    else if constexpr (MODE == M_ANGULAR) return (o & ~3) + v;
    else if constexpr (MODE == M_VRN2) {
        int b = o >> 2, uo = (o >> 1) & 1, vo = o & 1;
        return b * 16 + (uo * 2 + (v >> 1)) * 4 + vo * 2 + (v & 1);
    } else return o * 4 + v;   // VSNL / VRN3: views are b*4 + v
}

// One workgroup: 8 waves = 4 M-tiles (2 rows x 2 col-halves) x 2 N-slots.
// Each wave: one 16-pixel A fragment reused for two 16-channel B tiles.
// K = VIEWS*TAPS*CIN, chunked by 32, fully unrolled.
template <int MODE>
__global__ __launch_bounds__(256) void conv_wmma(
        const unsigned short* __restrict__ in,
        const unsigned short* __restrict__ wgt,
        unsigned short* __restrict__ outb,
        float* __restrict__ outf)
{
    constexpr int VIEWS = (MODE == M_SPATIAL) ? 1 : 4;
    constexpr int TAPS  = (MODE == M_ANGULAR) ? 1 : 9;
    constexpr int CIN   = (MODE == M_VRN2) ? 16 : 64;
    constexpr int KTOT  = VIEWS * TAPS * CIN;
    constexpr int SEG   = TAPS * CIN;
    constexpr bool F32OUT = (MODE == M_VSNL) || (MODE == M_VRN3);
    constexpr size_t IN_IMG = (size_t)98 * 98 * CIN;
    constexpr int ROW = 98 * CIN;

    const int outImg = blockIdx.y;
    const int h0 = (blockIdx.x / 3) * 2;
    const int w0 = (blockIdx.x % 3) * 32;
    const int lane = threadIdx.x & 31;
    const int wid  = threadIdx.x >> 5;
    const int mi = wid & 3;       // 4 M tiles: 2 rows x 2 col-halves
    const int ns = wid >> 2;      // 2 N slots (32 channels each)
    const int mr = mi >> 1, mc = mi & 1;
    const int lm = lane & 15;
    const int hi = lane >> 4;

    const int hrow = h0 + mr;
    const int wpix = w0 + mc * 16 + lm;               // A-row pixel (x coord)
    const int ncol = ns * 32 + lm;                    // first B column

    const unsigned short* wbase = wgt;
    if constexpr (MODE == M_ANGULAR)
        wbase += (size_t)(outImg & 3) * 64 * KTOT;
    const unsigned short* wrow1 = wbase + (size_t)ncol * KTOT;
    const unsigned short* wrow2 = wrow1 + (size_t)16 * KTOT;

    f32x8 acc1 = {0.f, 0.f, 0.f, 0.f, 0.f, 0.f, 0.f, 0.f};
    f32x8 acc2 = {0.f, 0.f, 0.f, 0.f, 0.f, 0.f, 0.f, 0.f};

#pragma unroll
    for (int kc = 0; kc < KTOT / 32; ++kc) {
        FragAB a, b1, b2;
        // A fragment: two 8-element K groups per ISA layout.  Under full
        // unroll (v,t) are wave-uniform compile-time constants and ci is
        // c_base + hi*8, so each access is one immediate-offset b128 load.
#pragma unroll
        for (int gsel = 0; gsel < 2; ++gsel) {
            int g  = kc * 32 + gsel * 16;             // hi*8 folded into ci
            int v  = g / SEG;
            int r  = g % SEG;
            int t  = r / CIN;
            int ci = (r % CIN) + hi * 8;
            int ty = (TAPS == 9) ? (t / 3) : 1;
            int tx = (TAPS == 9) ? (t % 3) : 1;
            size_t off = (size_t)in_img_idx<MODE>(outImg, v) * IN_IMG
                       + (size_t)(hrow + ty) * ROW
                       + (size_t)(wpix + tx) * CIN + ci;
            a.q[gsel] = *reinterpret_cast<const u32x4*>(in + off);
        }
        // B fragments: two 8-element K groups (hi*8, 16+hi*8) for fixed column.
        {
            const unsigned short* p1 = wrow1 + kc * 32 + hi * 8;
            const unsigned short* p2 = wrow2 + kc * 32 + hi * 8;
            b1.q[0] = *reinterpret_cast<const u32x4*>(p1);
            b1.q[1] = *reinterpret_cast<const u32x4*>(p1 + 16);
            b2.q[0] = *reinterpret_cast<const u32x4*>(p2);
            b2.q[1] = *reinterpret_cast<const u32x4*>(p2 + 16);
        }
        acc1 = __builtin_amdgcn_wmma_f32_16x16x32_bf16(
                false, a.v, false, b1.v, (short)0, acc1, false, false);
        acc2 = __builtin_amdgcn_wmma_f32_16x16x32_bf16(
                false, a.v, false, b2.v, (short)0, acc2, false, false);
    }

    if constexpr (F32OUT) {
        float* o = outf + (((size_t)outImg * 96 + hrow) * 96) * 64;
#pragma unroll
        for (int e = 0; e < 8; ++e) {
            int wc = w0 + mc * 16 + e + hi * 8;       // D: M = e + hi*8
            o[(size_t)wc * 64 + ns * 32 + lm]      = acc1[e];
            o[(size_t)wc * 64 + ns * 32 + 16 + lm] = acc2[e];
        }
    } else {
        unsigned short* o = outb + (size_t)outImg * (98 * 98 * 64)
                                 + (size_t)(hrow + 1) * (98 * 64);
#pragma unroll
        for (int e = 0; e < 8; ++e) {
            int wc = w0 + mc * 16 + e + hi * 8;
            o[(size_t)(wc + 1) * 64 + ns * 32 + lm]      = f2bf(acc1[e]);
            o[(size_t)(wc + 1) * 64 + ns * 32 + 16 + lm] = f2bf(acc2[e]);
        }
    }
}

// -------------------- initial conv (vsn0): C_in = 1 -----------------------
__global__ void k_init(const float* __restrict__ x, const float* __restrict__ w,
                       unsigned short* __restrict__ act)
{
    int idx = blockIdx.x * 256 + threadIdx.x;
    if (idx >= 16 * 9216) return;
    int o = idx / 9216, p = idx % 9216;
    int h = p / 96, wc = p % 96;
    int b = o >> 2, u = (o >> 1) & 1, vv = o & 1;

    float patch[4][9];
#pragma unroll
    for (int d = 0; d < 4; ++d)
#pragma unroll
        for (int t = 0; t < 9; ++t) {
            int sy = t / 3, sx = t % 3;
            int hh = h + sy - 1, ww = wc + sx - 1;
            float val = 0.f;
            if (hh >= 0 && hh < 96 && ww >= 0 && ww < 96)
                val = x[(((size_t)b * 96 + hh) * 96 + ww) * 4 + d];
            patch[d][t] = val;
        }

    unsigned short* op = act + (size_t)o * (98 * 98 * 64)
                             + ((size_t)(h + 1) * 98 + (wc + 1)) * 64;
    for (int co = 0; co < 64; ++co) {
        float s = 0.f;
#pragma unroll
        for (int d = 0; d < 4; ++d) {
            int i = (d >> 1) - u + 1, j = (d & 1) - vv + 1;
            const float* wp = w + co * 81 + i * 3 + j;
#pragma unroll
            for (int t = 0; t < 9; ++t)
                s += wp[(t / 3) * 27 + (t % 3) * 9] * patch[d][t];
        }
        op[co] = f2bf(s);
    }
}

// ------------------------- assemble / final --------------------------------
__device__ __forceinline__ int res_idx(int a) {
    if (a == 0 || a == 7 || a == 56 || a == 63) return -1;
    if (a < 7)  return a - 1;
    if (a < 56) return a - 2;
    return a - 3;
}
__device__ __forceinline__ int corner_off(int a) {
    return a == 0 ? 0 : (a == 7 ? 2 : (a == 56 ? 1 : 3));
}

__global__ void k_assemble(const float* __restrict__ vsn,
                           const float* __restrict__ x,
                           float* __restrict__ xcon)
{
    int idx = blockIdx.x * 256 + threadIdx.x;
    if (idx >= 4 * 9216) return;
    const float* v  = vsn + (size_t)idx * 64;
    const float* xp = x   + (size_t)idx * 4;
    float* o = xcon + (size_t)idx * 64;
    for (int a = 0; a < 64; ++a) {
        int r = res_idx(a);
        o[a] = (r < 0) ? xp[corner_off(a)] : v[r];
    }
}

__global__ void k_final(const float* __restrict__ vsn,
                        const float* __restrict__ vrn,
                        const float* __restrict__ x,
                        float* __restrict__ out)
{
    int idx = blockIdx.x * 256 + threadIdx.x;
    if (idx >= 4 * 9216) return;
    const float* a1 = vsn + (size_t)idx * 64;
    const float* a2 = vrn + (size_t)idx * 64;
    const float* xp = x   + (size_t)idx * 4;
    float* o = out + (size_t)idx * 64;
    for (int a = 0; a < 64; ++a) {
        int r = res_idx(a);
        o[a] = (r < 0) ? xp[corner_off(a)] : (a1[r] + a2[r]);
    }
}

// -------------------- vrn1: C_in = 1, 8x8 -> 4x4, stride 2 ----------------
__global__ void k_vrn1(const float* __restrict__ xcon,
                       const float* __restrict__ w,
                       unsigned short* __restrict__ g1)
{
    int idx = blockIdx.x * 256 + threadIdx.x;
    if (idx >= 4 * 16 * 9216) return;
    int p = idx % 9216;
    int view = (idx / 9216) % 16;
    int b = idx / (9216 * 16);
    int h = p / 96, wc = p % 96;
    int uo = view >> 2, vo = view & 3;

    float patch[4][9];
#pragma unroll
    for (int d = 0; d < 4; ++d) {
        int a = (2 * uo + (d >> 1)) * 8 + (2 * vo + (d & 1));
#pragma unroll
        for (int t = 0; t < 9; ++t) {
            int hh = h + t / 3 - 1, ww = wc + t % 3 - 1;
            float val = 0.f;
            if (hh >= 0 && hh < 96 && ww >= 0 && ww < 96)
                val = xcon[(((size_t)b * 96 + hh) * 96 + ww) * 64 + a];
            patch[d][t] = val;
        }
    }
    unsigned short* op = g1 + (size_t)(b * 16 + view) * (98 * 98 * 16)
                            + ((size_t)(h + 1) * 98 + (wc + 1)) * 16;
    for (int co = 0; co < 16; ++co) {
        float s = 0.f;
#pragma unroll
        for (int d = 0; d < 4; ++d)
#pragma unroll
            for (int t = 0; t < 9; ++t)
                s += w[co * 36 + (t / 3) * 12 + (t % 3) * 4 + d] * patch[d][t];
        op[co] = f2bf(s);
    }
}

// -------------------------- weight repack kernels --------------------------
// Wsp[l][co][k = t*64+ci]
__global__ void k_repack_sp(const float* __restrict__ w, unsigned short* __restrict__ o) {
    int idx = blockIdx.x * 256 + threadIdx.x;
    if (idx >= 16 * 64 * 576) return;
    int l = idx / 36864, r = idx % 36864;
    int co = r / 576, k = r % 576;
    int t = k / 64, ci = k % 64;
    o[idx] = f2bf(w[((((size_t)l * 64 + co) * 64 + ci) * 3 + t / 3) * 3 + t % 3]);
}
// Wang[l][vo4][co][k = vi4*64+ci]
__global__ void k_repack_ang(const float* __restrict__ w, unsigned short* __restrict__ o) {
    int idx = blockIdx.x * 256 + threadIdx.x;
    if (idx >= 16 * 4 * 64 * 256) return;
    int l = idx / 65536, r = idx % 65536;
    int vo4 = r / 16384, r2 = r % 16384;
    int co = r2 / 256, k = r2 % 256;
    int vi4 = k >> 6, ci = k & 63;
    int i = (vi4 >> 1) - (vo4 >> 1) + 1;
    int j = (vi4 & 1) - (vo4 & 1) + 1;
    o[idx] = f2bf(w[((((size_t)l * 64 + co) * 64 + ci) * 3 + i) * 3 + j]);
}
// W[co<64 pad][k = (d*9+t)*64+ci]  (vsn_last & vrn3, shape [60][64][3][3][2][2])
__global__ void k_repack_corner(const float* __restrict__ w, unsigned short* __restrict__ o) {
    int idx = blockIdx.x * 256 + threadIdx.x;
    if (idx >= 64 * 2304) return;
    int co = idx / 2304, k = idx % 2304;
    int d = k / 576, t = (k / 64) % 9, ci = k % 64;
    float val = 0.f;
    if (co < 60)
        val = w[(((((size_t)co * 64 + ci) * 3 + t / 3) * 3 + t % 3) * 2 + (d >> 1)) * 2 + (d & 1)];
    o[idx] = f2bf(val);
}
// Wvrn2[co][k = (d*9+t)*16+ci]
__global__ void k_repack_vrn2(const float* __restrict__ w, unsigned short* __restrict__ o) {
    int idx = blockIdx.x * 256 + threadIdx.x;
    if (idx >= 64 * 576) return;
    int co = idx / 576, k = idx % 576;
    int d = k / 144, t = (k / 16) % 9, ci = k % 16;
    o[idx] = f2bf(w[(((((size_t)co * 16 + ci) * 3 + t / 3) * 3 + t % 3) * 2 + (d >> 1)) * 2 + (d & 1)]);
}

// ---------------------------------------------------------------------------
extern "C" void kernel_launch(void* const* d_in, const int* in_sizes, int n_in,
                              void* d_out, int out_size, void* d_ws, size_t ws_size,
                              hipStream_t stream)
{
    (void)in_sizes; (void)n_in; (void)out_size; (void)ws_size;
    const float* x      = (const float*)d_in[0];
    const float* w_vsn0 = (const float*)d_in[1];
    const float* w_s    = (const float*)d_in[2];
    const float* w_a    = (const float*)d_in[3];
    const float* w_vsnl = (const float*)d_in[4];
    const float* w_vrn1 = (const float*)d_in[5];
    const float* w_vrn2 = (const float*)d_in[6];
    const float* w_vrn3 = (const float*)d_in[7];

    char* ws = (char*)d_ws;
    const size_t ACT = (size_t)16 * 98 * 98 * 64 * 2;      // 19,668,992 B
    const size_t VSN = (size_t)4 * 96 * 96 * 64 * 4;       //  9,437,184 B
    unsigned short* actA  = (unsigned short*)(ws);
    unsigned short* actB  = (unsigned short*)(ws + ACT);
    unsigned short* g1    = (unsigned short*)(ws + 2 * ACT);
    unsigned short* g2    = (unsigned short*)(ws + 3 * ACT);
    float*          x_vsn = (float*)(ws + 4 * ACT);
    float*          x_vrn = (float*)(ws + 4 * ACT + VSN);
    float*          x_con = (float*)(ws + 4 * ACT + 2 * VSN);
    char* wbase = ws + 4 * ACT + 3 * VSN;
    unsigned short* Wsp   = (unsigned short*)(wbase);
    unsigned short* Wang  = (unsigned short*)(wbase + 1179648);
    unsigned short* Wvsnl = (unsigned short*)(wbase + 1179648 + 2097152);
    unsigned short* Wvrn2 = (unsigned short*)(wbase + 1179648 + 2097152 + 294912);
    unsigned short* Wvrn3 = (unsigned short*)(wbase + 1179648 + 2097152 + 294912 + 73728);

    // Zero padded activation buffers (halos stay zero for the whole launch).
    hipMemsetAsync(ws, 0, 4 * ACT, stream);

    // Weight repack (fp32 -> bf16, GEMM-friendly [co][k] layout).
    k_repack_sp    <<<2304, 256, 0, stream>>>(w_s,    Wsp);
    k_repack_ang   <<<4096, 256, 0, stream>>>(w_a,    Wang);
    k_repack_corner<<< 576, 256, 0, stream>>>(w_vsnl, Wvsnl);
    k_repack_vrn2  <<< 144, 256, 0, stream>>>(w_vrn2, Wvrn2);
    k_repack_corner<<< 576, 256, 0, stream>>>(w_vrn3, Wvrn3);

    // Initial 4D conv (C_in = 1).
    k_init<<<576, 256, 0, stream>>>(x, w_vsn0, actA);

    // 16 x (spatial WMMA conv, angular WMMA conv), ping-pong in L2.
    unsigned short* cur = actA;
    unsigned short* nxt = actB;
    for (int l = 0; l < 16; ++l) {
        conv_wmma<M_SPATIAL><<<dim3(144, 16), 256, 0, stream>>>(
            cur, Wsp + (size_t)l * 64 * 576, nxt, nullptr);
        { unsigned short* t = cur; cur = nxt; nxt = t; }
        conv_wmma<M_ANGULAR><<<dim3(144, 16), 256, 0, stream>>>(
            cur, Wang + (size_t)l * 4 * 64 * 256, nxt, nullptr);
        { unsigned short* t = cur; cur = nxt; nxt = t; }
    }

    // vsn_last: 4 views x 9 taps x 64 -> 60(64) channels, f32 out.
    conv_wmma<M_VSNL><<<dim3(144, 4), 256, 0, stream>>>(cur, Wvsnl, nullptr, x_vsn);

    // Assemble 8x8 grid, then vrn chain.
    k_assemble<<<144, 256, 0, stream>>>(x_vsn, x, x_con);
    k_vrn1<<<2304, 256, 0, stream>>>(x_con, w_vrn1, g1);
    conv_wmma<M_VRN2><<<dim3(144, 16), 256, 0, stream>>>(g1, Wvrn2, g2, nullptr);
    conv_wmma<M_VRN3><<<dim3(144, 4), 256, 0, stream>>>(g2, Wvrn3, nullptr, x_vrn);

    // Final assemble: out = assemble(x_vsn + x_vrn, x).
    k_final<<<144, 256, 0, stream>>>(x_vsn, x_vrn, x, (float*)d_out);
}
